// TransitionDown_37795712204991
// MI455X (gfx1250) — matrix operand order repaired and checked by reading, not compiled, not run on hardware
//
#include <hip/hip_runtime.h>
#include <hip/hip_bf16.h>

typedef __attribute__((ext_vector_type(16))) _Float16 v16h;
typedef __attribute__((ext_vector_type(8)))  float    v8f;

#define BB 4
#define NN 16384
#define DD 61
#define SS 4096
#define KS 16
#define C0 64
#define C1 128
#define C2 128
#define BN_EPSF 1e-5f

// ---------------------------------------------------------------------------
// Stage 1: farthest point sampling. One 1024-thread workgroup per batch.
// xyz (192KB) + running dist (64KB) live in LDS for the whole 4096-step loop.
// Writes new_xyz directly into d_out[0 .. B*S*3).
// ---------------------------------------------------------------------------
__global__ __launch_bounds__(1024) void fps_kernel(const float* __restrict__ xyz,
                                                   float* __restrict__ out_xyz) {
  extern __shared__ float dyn[];
  float* sx    = dyn;            // NN*3 floats
  float* sdist = dyn + NN * 3;   // NN floats
  __shared__ float redv[1024];
  __shared__ int   redi[1024];

  const int b   = blockIdx.x;
  const int tid = threadIdx.x;
  const float* xb = xyz + (size_t)b * NN * 3;

  for (int k = tid; k < NN * 3; k += 1024) sx[k] = xb[k];
  for (int n = tid; n < NN; n += 1024) sdist[n] = 1e10f;
  __syncthreads();

  int far = 0;
  for (int i = 0; i < SS; ++i) {
    const float cx = sx[far * 3 + 0];
    const float cy = sx[far * 3 + 1];
    const float cz = sx[far * 3 + 2];
    if (tid < 3) out_xyz[((size_t)b * SS + i) * 3 + tid] = (tid == 0) ? cx : ((tid == 1) ? cy : cz);

    float bv = -1.0f; int bi = 0;
    for (int n = tid; n < NN; n += 1024) {
      float dx = sx[n * 3 + 0] - cx;
      float dy = sx[n * 3 + 1] - cy;
      float dz = sx[n * 3 + 2] - cz;
      float d  = dx * dx + dy * dy + dz * dz;
      float nd = fminf(sdist[n], d);
      sdist[n] = nd;
      if (nd > bv) { bv = nd; bi = n; }   // strict > keeps smallest n on ties
    }
    redv[tid] = bv; redi[tid] = bi;
    for (int off = 512; off > 0; off >>= 1) {
      __syncthreads();
      if (tid < off) {
        float v2 = redv[tid + off]; int i2 = redi[tid + off];
        if (v2 > redv[tid] || (v2 == redv[tid] && i2 < redi[tid])) { redv[tid] = v2; redi[tid] = i2; }
      }
    }
    __syncthreads();
    far = redi[0];
    __syncthreads();
  }
}

// ---------------------------------------------------------------------------
// Stage 2: 16-NN per centroid. One wave32 per centroid (8 waves / block).
// Per-lane sorted top-16 over 512 strided points, then 32-way sorted-list
// merge with shfl_xor wave reductions.
// ---------------------------------------------------------------------------
__global__ __launch_bounds__(256) void knn_kernel(const float* __restrict__ xyz,
                                                  const float* __restrict__ new_xyz,
                                                  int* __restrict__ knn_idx) {
  __shared__ float sd[8 * 512];
  __shared__ int   sn[8 * 512];
  const int tid  = threadIdx.x;
  const int w    = tid >> 5;
  const int lane = tid & 31;
  const int g    = blockIdx.x * 8 + w;   // global centroid id = b*SS + s
  const int b    = g / SS;

  const float cx = new_xyz[(size_t)g * 3 + 0];
  const float cy = new_xyz[(size_t)g * 3 + 1];
  const float cz = new_xyz[(size_t)g * 3 + 2];

  float bd[KS]; int bi[KS];
#pragma unroll
  for (int j = 0; j < KS; ++j) { bd[j] = 3.0e38f; bi[j] = 0x7fffffff; }

  const float* xb = xyz + (size_t)b * NN * 3;
  for (int n = lane; n < NN; n += 32) {
    float dx = xb[n * 3 + 0] - cx;
    float dy = xb[n * 3 + 1] - cy;
    float dz = xb[n * 3 + 2] - cz;
    float d  = dx * dx + dy * dy + dz * dz;
    if (d < bd[KS - 1]) {
      float cdv = d; int civ = n;
#pragma unroll
      for (int j = 0; j < KS; ++j) {
        if (cdv < bd[j]) { float t = bd[j]; int ti = bi[j]; bd[j] = cdv; bi[j] = civ; cdv = t; civ = ti; }
      }
    }
  }
#pragma unroll
  for (int j = 0; j < KS; ++j) { sd[w * 512 + lane * 16 + j] = bd[j]; sn[w * 512 + lane * 16 + j] = bi[j]; }
  __syncthreads();

  int head = 0;
  for (int r = 0; r < KS; ++r) {
    float v  = (head < KS) ? sd[w * 512 + lane * 16 + head] : 3.4e38f;
    int   ni = (head < KS) ? sn[w * 512 + lane * 16 + head] : 0x7fffffff;
    int   src = lane;
#pragma unroll
    for (int off = 16; off > 0; off >>= 1) {
      float v2 = __shfl_xor(v, off, 32);
      int   n2 = __shfl_xor(ni, off, 32);
      int   s2 = __shfl_xor(src, off, 32);
      if (v2 < v || (v2 == v && n2 < ni)) { v = v2; ni = n2; src = s2; }
    }
    if (lane == src) head++;
    if (lane == 0) knn_idx[(size_t)g * KS + r] = ni;
  }
}

// ---------------------------------------------------------------------------
// Swizzled-fragment LDS layout: each (fragment, lane) owns 16 contiguous
// halfs (32B), so a WMMA operand load is one 32B vector LDS read.
//
// A (16x32 MxK f16) ISA layout: lane = m + 16*hi where hi selects K sub-
// blocks {8..15,24..31}; reg pair j covers K = (j&4?16:0)+hi*8+2*(j&3)+{0,1}.
// B (32x16 KxN f16): lane = n + 16*hi, hi = (k>>3)&1 (same K order as A),
// pos = ((k>>4)&1)*8 + (k&7).
// ---------------------------------------------------------------------------
__device__ inline int a_swz_idx(int frag, int m, int kprime) {
  int hi   = (kprime >> 3) & 1;
  int lane = m + (hi << 4);
  int j    = (((kprime >> 4) & 1) << 2) | ((kprime & 7) >> 1);
  int pos  = (j << 1) | (kprime & 1);
  return ((frag * 32 + lane) << 4) + pos;
}

__device__ inline int b_swz_idx(int frag, int n, int kprime) {
  int hi   = (kprime >> 3) & 1;
  int lane = n + (hi << 4);
  return ((frag * 32 + lane) << 4) + ((((kprime >> 4) & 1) << 3) | (kprime & 7));
}

__device__ inline v16h load_frag16(const _Float16* base, int frag, int lane) {
  return *(const v16h*)(base + ((frag * 32 + lane) << 4));   // 32B -> 2x ds_load_b128
}

// ---------------------------------------------------------------------------
// Stage 3: gather + conv0(bn,relu) + conv1(bn,relu) + max over k=16.
// Block = 16 centroids -> M=256 rows; 8 waves, wave w owns channel tile w.
// A0/H/weights staged in LDS pre-swizzled to fragment layout; WMMA f32
// accumulate; BN folded into scale/shift; maxpool via shfl_xor(16).
// ---------------------------------------------------------------------------
__global__ __launch_bounds__(256) void group_gemm_kernel(
    const float* __restrict__ xyz, const float* __restrict__ points,
    const float* __restrict__ w0, const float* __restrict__ b0,
    const float* __restrict__ g0, const float* __restrict__ be0,
    const float* __restrict__ m0, const float* __restrict__ v0,
    const float* __restrict__ w1, const float* __restrict__ b1,
    const float* __restrict__ g1, const float* __restrict__ be1,
    const float* __restrict__ m1, const float* __restrict__ v1,
    const int* __restrict__ knn_idx, const float* __restrict__ new_xyz,
    float* __restrict__ out_pts) {
  extern __shared__ char smem[];
  _Float16* A0s = (_Float16*)smem;        // 16 mtiles * 2 kc * 512 = 256x64
  _Float16* Hs  = A0s + 256 * C0;         // 16 mtiles * 4 kc * 512 = 256x128
  _Float16* W0s = Hs + 256 * C1;          // 8 ntiles * 2 kc * 512 = 128x64
  _Float16* W1s = W0s + C1 * C0;          // 8 ntiles * 4 kc * 512 = 128x128
  float* sc0 = (float*)(W1s + C2 * C1);   // 128
  float* sh0 = sc0 + C1;
  float* sc1 = sh0 + C1;
  float* sh1 = sc1 + C2;

  const int tid  = threadIdx.x;
  const int w    = tid >> 5;
  const int lane = tid & 31;
  const int blk  = blockIdx.x;
  const int b    = blk >> 8;              // SS/16 = 256 tiles per batch
  const int s0   = (blk & 255) << 4;

  if (tid < C1) {
    float s = g0[tid] * rsqrtf(v0[tid] + BN_EPSF);
    sc0[tid] = s;
    sh0[tid] = (b0[tid] - m0[tid]) * s + be0[tid];
    float t = g1[tid] * rsqrtf(v1[tid] + BN_EPSF);
    sc1[tid] = t;
    sh1[tid] = (b1[tid] - m1[tid]) * t + be1[tid];
  }
  // stage weights as f16 in swizzled B-fragment layout (written once, read
  // as single b128 vector loads by every wave)
  for (int idx = tid; idx < C1 * C0; idx += 256) {
    int o = idx >> 6, c = idx & 63;
    W0s[b_swz_idx((o >> 4) * 2 + (c >> 5), o & 15, c & 31)] = (_Float16)w0[idx];
  }
  for (int idx = tid; idx < C2 * C1; idx += 256) {
    int o = idx >> 7, c = idx & 127;
    W1s[b_swz_idx((o >> 4) * 4 + (c >> 5), o & 15, c & 31)] = (_Float16)w1[idx];
  }

  { // gather one row (centroid i, neighbor kn) per thread, scatter into
    // swizzled A-fragment layout
    const int i  = tid >> 4, kn = tid & 15;
    const int mt = i, m = kn;
    const int s  = s0 + i;
    const int n  = knn_idx[((size_t)(b * SS + s)) * KS + kn];
    const float* cen = new_xyz + (size_t)(b * SS + s) * 3;
    const float* px  = xyz + ((size_t)b * NN + n) * 3;
    A0s[a_swz_idx(mt * 2 + 0, m, 0)] = (_Float16)(px[0] - cen[0]);
    A0s[a_swz_idx(mt * 2 + 0, m, 1)] = (_Float16)(px[1] - cen[1]);
    A0s[a_swz_idx(mt * 2 + 0, m, 2)] = (_Float16)(px[2] - cen[2]);
    const float* pp = points + ((size_t)b * NN + n) * DD;
#pragma unroll 1
    for (int c = 3; c < C0; ++c)
      A0s[a_swz_idx(mt * 2 + (c >> 5), m, c & 31)] = (_Float16)pp[c - 3];
  }
  __syncthreads();

  const int nloc = lane & 15;
  const int hi   = (lane >> 4) & 1;
  const int ch   = w * 16 + nloc;         // this lane's output channel
  const int kc1  = ch >> 5;               // layer-1 k-chunk this channel is in
  const int kp1  = ch & 31;               // k' within that chunk

  { // layer 0: M=256, K=64, N=128; B frags in regs across all M tiles
    v16h bf0 = load_frag16(W0s, w * 2 + 0, lane);
    v16h bf1 = load_frag16(W0s, w * 2 + 1, lane);
    const float sc = sc0[ch], sh = sh0[ch];
    for (int mt = 0; mt < 16; ++mt) {
      v8f acc = {};
      v16h a0f = load_frag16(A0s, mt * 2 + 0, lane);
      acc = __builtin_amdgcn_wmma_f32_16x16x32_f16(false, a0f, false, bf0, (short)0, acc, false, false);
      v16h a1f = load_frag16(A0s, mt * 2 + 1, lane);
      acc = __builtin_amdgcn_wmma_f32_16x16x32_f16(false, a1f, false, bf1, (short)0, acc, false, false);
#pragma unroll
      for (int r = 0; r < 8; ++r) {
        int m = r + (hi << 3);            // output row within tile mt
        float y = fmaxf(acc[r] * sc + sh, 0.0f);
        Hs[a_swz_idx(mt * 4 + kc1, m, kp1)] = (_Float16)y;
      }
    }
  }
  __syncthreads();

  { // layer 1: M=256, K=128, N=128; fused bn+relu+max-over-k, direct store
    v16h bf[4];
#pragma unroll
    for (int kc = 0; kc < 4; ++kc) bf[kc] = load_frag16(W1s, w * 4 + kc, lane);
    const float sc = sc1[ch], sh = sh1[ch];
    for (int mt = 0; mt < 16; ++mt) {
      v8f acc = {};
#pragma unroll
      for (int kc = 0; kc < 4; ++kc) {
        v16h af = load_frag16(Hs, mt * 4 + kc, lane);
        acc = __builtin_amdgcn_wmma_f32_16x16x32_f16(false, af, false, bf[kc], (short)0, acc, false, false);
      }
      float vm = 0.0f;   // relu output is >= 0
#pragma unroll
      for (int r = 0; r < 8; ++r) vm = fmaxf(vm, fmaxf(acc[r] * sc + sh, 0.0f));
      vm = fmaxf(vm, __shfl_xor(vm, 16, 32));   // combine row halves r / r+8
      if (lane < 16) out_pts[((size_t)(b * SS + s0 + mt)) * C2 + ch] = vm;
    }
  }
}

// ---------------------------------------------------------------------------
extern "C" void kernel_launch(void* const* d_in, const int* in_sizes, int n_in,
                              void* d_out, int out_size, void* d_ws, size_t ws_size,
                              hipStream_t stream) {
  const float* xyz    = (const float*)d_in[0];
  const float* points = (const float*)d_in[1];
  const float* w0 = (const float*)d_in[2];
  const float* b0 = (const float*)d_in[3];
  const float* g0 = (const float*)d_in[4];
  const float* be0 = (const float*)d_in[5];
  const float* m0 = (const float*)d_in[6];
  const float* v0 = (const float*)d_in[7];
  const float* w1 = (const float*)d_in[8];
  const float* b1 = (const float*)d_in[9];
  const float* g1 = (const float*)d_in[10];
  const float* be1 = (const float*)d_in[11];
  const float* m1 = (const float*)d_in[12];
  const float* v1 = (const float*)d_in[13];

  float* out      = (float*)d_out;
  float* new_xyz  = out;                     // B*S*3 floats
  float* out_pts  = out + (size_t)BB * SS * 3;
  int*   knn      = (int*)d_ws;              // B*S*16 ints

  const size_t fps_lds  = (size_t)(NN * 3 + NN) * sizeof(float);          // 256 KB (<=320KB WGP LDS)
  const size_t gemm_lds = (size_t)(256 * C0 + 256 * C1 + C1 * C0 + C2 * C1) * sizeof(_Float16)
                        + (size_t)(4 * C1) * sizeof(float);               // ~146 KB

  (void)hipFuncSetAttribute((const void*)fps_kernel,
                            hipFuncAttributeMaxDynamicSharedMemorySize, (int)fps_lds);
  (void)hipFuncSetAttribute((const void*)group_gemm_kernel,
                            hipFuncAttributeMaxDynamicSharedMemorySize, (int)gemm_lds);

  fps_kernel<<<BB, 1024, fps_lds, stream>>>(xyz, out);
  knn_kernel<<<(BB * SS) / 8, 256, 0, stream>>>(xyz, new_xyz, knn);
  group_gemm_kernel<<<BB * (SS / 16), 256, gemm_lds, stream>>>(
      xyz, points, w0, b0, g0, be0, m0, v0, w1, b1, g1, be1, m1, v1,
      knn, new_xyz, out_pts);
}
